// TripletLoss_30717606101531
// MI455X (gfx1250) — matrix operand order repaired and checked by reading, not compiled, not run on hardware
//
#include <hip/hip_runtime.h>
#include <math.h>

// ---------------------------------------------------------------------------
// TripletLoss on MI455X (gfx1250): fused GEMM + row-min using
// v_wmma_f32_16x16x32_bf16 (bf16x3 error-compensated split, ~fp32 accuracy),
// with double-buffered GLOBAL_LOAD_ASYNC_TO_LDS_B128 staging of pre-split
// bf16 B tiles (ASYNCcnt path). wave32, no atomics (deterministic).
// ---------------------------------------------------------------------------

typedef __attribute__((ext_vector_type(16))) __bf16          v16bf;
typedef __attribute__((ext_vector_type(16))) unsigned short  v16u;
typedef __attribute__((ext_vector_type(8)))  unsigned short  v8u;
typedef __attribute__((ext_vector_type(4)))  unsigned short  v4u;
typedef __attribute__((ext_vector_type(8)))  float           v8f;
typedef __attribute__((ext_vector_type(4)))  float           v4f;
typedef __attribute__((ext_vector_type(4)))  int             v4i;

#define BS          8192            // batch split size
#define D           128             // feature dim
#define NSPLIT      4               // N-dimension grid split (partial mins)
#define LDS_STRIDE  144             // 128 + 16 bf16 pad (row stride, bank stagger)

// address-space qualified 16-byte vector types for the async-copy builtin:
// signature is (v4i addrspace(1)*, v4i addrspace(3)*, imm offset, imm cpol)
typedef __attribute__((address_space(1))) v4i as1_v4i;
typedef __attribute__((address_space(3))) v4i as3_v4i;

__device__ __forceinline__ unsigned short f2bf_hi(float f) {
    // round-to-nearest-even fp32 -> bf16 (inputs are finite normals)
    unsigned u = __float_as_uint(f);
    u += 0x7FFFu + ((u >> 16) & 1u);
    return (unsigned short)(u >> 16);
}
__device__ __forceinline__ float bf2f(unsigned short h) {
    return __uint_as_float(((unsigned)h) << 16);
}

// ---- CDNA5 async copy global->LDS (ASYNCcnt) ------------------------------
__device__ __forceinline__ void async_copy_b128(const unsigned short* g,
                                                unsigned short* l) {
#if __has_builtin(__builtin_amdgcn_global_load_async_to_lds_b128)
    __builtin_amdgcn_global_load_async_to_lds_b128(
        (as1_v4i*)g, (as3_v4i*)l, 0, 0);
#else
    unsigned loff = (unsigned)(unsigned long long)(as3_v4i*)l;
    asm volatile("global_load_async_to_lds_b128 %0, %1, off"
                 :: "v"(loff), "v"((unsigned long long)g) : "memory");
#endif
}

#if __has_builtin(__builtin_amdgcn_s_wait_asynccnt)
#define WAIT_ASYNC(n) __builtin_amdgcn_s_wait_asynccnt(n)
#else
#define WAIT_ASYNC(n) asm volatile("s_wait_asynccnt %0" :: "i"(n) : "memory")
#endif

// ---------------------------------------------------------------------------
// Kernel 1: per-row squared norms, exact fp32 paired distance, and the bf16
// hi/lo split of the positive rows (so the GEMM kernel can async-copy B tiles
// straight to LDS with no per-tile conversion).
// One wave32 per row-pair i.
// ---------------------------------------------------------------------------
__global__ __launch_bounds__(256) void prep_kernel(const float* __restrict__ feats,
                                                   float* __restrict__ sq_a,
                                                   float* __restrict__ sq_p,
                                                   float* __restrict__ pd,
                                                   unsigned short* __restrict__ pos_hi,
                                                   unsigned short* __restrict__ pos_lo) {
    const int i    = (blockIdx.x * blockDim.x + threadIdx.x) >> 5;
    const int lane = threadIdx.x & 31;
    if (i >= BS) return;
    const float* arow = feats + (size_t)i * D;
    const float* prow = feats + (size_t)(i + BS) * D;
    v4f a = *(const v4f*)(arow + lane * 4);
    v4f p = *(const v4f*)(prow + lane * 4);
    float asq = 0.f, psq = 0.f, dsq = 0.f;
    v4u h4, l4;
#pragma unroll
    for (int e = 0; e < 4; ++e) {
        asq = fmaf(a[e], a[e], asq);
        psq = fmaf(p[e], p[e], psq);
        float d = a[e] - p[e];
        dsq = fmaf(d, d, dsq);
        unsigned short h = f2bf_hi(p[e]);
        h4[e] = h;
        l4[e] = f2bf_hi(p[e] - bf2f(h));
    }
    *(v4u*)&pos_hi[(size_t)i * D + lane * 4] = h4;
    *(v4u*)&pos_lo[(size_t)i * D + lane * 4] = l4;
#pragma unroll
    for (int m = 16; m >= 1; m >>= 1) {
        asq += __shfl_xor(asq, m, 32);
        psq += __shfl_xor(psq, m, 32);
        dsq += __shfl_xor(dsq, m, 32);
    }
    if (lane == 0) {
        sq_a[i] = asq;
        sq_p[i] = psq;
        pd[i]   = sqrtf(dsq);
    }
}

// ---------------------------------------------------------------------------
// Kernel 2: fused cross-GEMM + per-row min(d2) over an N-split.
// grid = (BS/128, NSPLIT), block = 256 threads = 8 waves; wave w owns one
// 16-row M-tile. B tiles double-buffered in LDS via async global->LDS copies.
// 12 WMMAs per 16x16x128 output tile (bf16x3 split).
// ---------------------------------------------------------------------------
__global__ __launch_bounds__(256) void tripmin_kernel(const float* __restrict__ feats,
                                                      const float* __restrict__ sq_a,
                                                      const float* __restrict__ sq_p,
                                                      const unsigned short* __restrict__ pos_hi,
                                                      const unsigned short* __restrict__ pos_lo,
                                                      float* __restrict__ minpart) {
    // [buffer][hi/lo][16 rows * LDS_STRIDE]
    __shared__ __align__(16) unsigned short ldsBuf[2][2][16 * LDS_STRIDE];

    const int tid  = threadIdx.x;
    const int lane = tid & 31;
    const int wave = tid >> 5;
    const int m0   = blockIdx.x * 128 + wave * 16;
    const int mrow = m0 + (lane & 15);
    const int ksel = (lane >> 4) & 1;           // which half of the K-run this lane holds

    // staging map: thread t copies one 16B chunk of hi and one of lo
    const int srow = tid >> 4;                  // 0..15  (tile row)
    const int sch  = tid & 15;                  // 0..15  (8-element chunk)
    const int soff = srow * LDS_STRIDE + sch * 8;

    // ---- cache A fragments (hi/lo bf16) for all 4 K-chunks ---------------
    // 16-bit A 16x32 layout: lanes 0-15 hold K = {c..c+7, c+16..c+23},
    //                        lanes 16-31 hold K = {c+8..c+15, c+24..c+31}.
    v16bf Ahi[4], Alo[4];
#pragma unroll
    for (int kc = 0; kc < 4; ++kc) {
        const float* base = feats + (size_t)mrow * D + kc * 32 + ksel * 8;
        v16u hu, lu;
#pragma unroll
        for (int half = 0; half < 2; ++half) {      // +0 and +16 K offset
            v4f f0 = *(const v4f*)(base + half * 16);
            v4f f1 = *(const v4f*)(base + half * 16 + 4);
#pragma unroll
            for (int e = 0; e < 4; ++e) {
                unsigned short h0 = f2bf_hi(f0[e]);
                unsigned short h1 = f2bf_hi(f1[e]);
                hu[half * 8 + e]     = h0;
                lu[half * 8 + e]     = f2bf_hi(f0[e] - bf2f(h0));
                hu[half * 8 + 4 + e] = h1;
                lu[half * 8 + 4 + e] = f2bf_hi(f1[e] - bf2f(h1));
            }
        }
        Ahi[kc] = __builtin_bit_cast(v16bf, hu);
        Alo[kc] = __builtin_bit_cast(v16bf, lu);
    }

    // f32 C/D layout: VGPR r, lanes 0-15 -> M=r, N=lane; lanes 16-31 -> M=8+r.
    float asq_r[8];
#pragma unroll
    for (int r = 0; r < 8; ++r) asq_r[r] = sq_a[m0 + r + ksel * 8];

    float minD2[8];
#pragma unroll
    for (int r = 0; r < 8; ++r) minD2[r] = 3.4e38f;

    const int ntiles = (BS / 16) / NSPLIT;          // 128 tiles per split
    const int ntbase = blockIdx.y * ntiles;
    const int brow   = lane & 15;                   // B column / tile row

    // ---- prologue: async-stage tile 0 into buffer 0 ----------------------
    {
        const size_t g = (size_t)(ntbase * 16 + srow) * D + sch * 8;
        async_copy_b128(pos_hi + g, &ldsBuf[0][0][soff]);
        async_copy_b128(pos_lo + g, &ldsBuf[0][1][soff]);
    }

    for (int t = 0; t < ntiles; ++t) {
        const int cur = t & 1;
        const int n0  = (ntbase + t) * 16;

        if (t + 1 < ntiles) {
            // async-stage next tile into the other buffer (safe: it was
            // fully consumed before the trailing barrier of iteration t-1)
            const size_t g = (size_t)(n0 + 16 + srow) * D + sch * 8;
            async_copy_b128(pos_hi + g, &ldsBuf[cur ^ 1][0][soff]);
            async_copy_b128(pos_lo + g, &ldsBuf[cur ^ 1][1][soff]);
            WAIT_ASYNC(2);                  // tile t's 2 copies (in-order) done
        } else {
            WAIT_ASYNC(0);
        }
        __syncthreads();                    // all waves' copies of tile t visible

        const unsigned short* ldsHi = &ldsBuf[cur][0][0];
        const unsigned short* ldsLo = &ldsBuf[cur][1][0];

        // ---- 16x16x128 cross tile via 12 bf16 WMMAs (bf16x3 split) -------
        v8f c = {};
#pragma unroll
        for (int kc = 0; kc < 4; ++kc) {
            // 16-bit B 32x16 layout: lane n<16 = column n, K = {c..c+7, c+16..c+23};
            //                        lane n+16 = column n, K = {c+8..c+15, c+24..c+31}.
            const int off = brow * LDS_STRIDE + kc * 32 + ksel * 8;
            const v8u* ph = (const v8u*)&ldsHi[off];
            const v8u* pl = (const v8u*)&ldsLo[off];
            v8u h0 = ph[0], h1 = ph[2], l0 = pl[0], l1 = pl[2];
            v16u bh, bl;
#pragma unroll
            for (int e = 0; e < 8; ++e) {
                bh[e] = h0[e]; bh[8 + e] = h1[e];
                bl[e] = l0[e]; bl[8 + e] = l1[e];
            }
            v16bf Bhi = __builtin_bit_cast(v16bf, bh);
            v16bf Blo = __builtin_bit_cast(v16bf, bl);
            c = __builtin_amdgcn_wmma_f32_16x16x32_bf16(false, Alo[kc], false, Bhi,
                                                        (short)0, c, false, false);
            c = __builtin_amdgcn_wmma_f32_16x16x32_bf16(false, Ahi[kc], false, Blo,
                                                        (short)0, c, false, false);
            c = __builtin_amdgcn_wmma_f32_16x16x32_bf16(false, Ahi[kc], false, Bhi,
                                                        (short)0, c, false, false);
        }

        const float psq = sq_p[n0 + brow];
#pragma unroll
        for (int r = 0; r < 8; ++r) {
            float d2 = fmaf(-2.0f, c[r], asq_r[r] + psq);
            minD2[r] = fminf(minD2[r], d2);
        }

        __syncthreads();                    // tile t fully read by all waves
    }

    // ---- min across the 16 columns held by each 16-lane half -------------
#pragma unroll
    for (int r = 0; r < 8; ++r) {
        float v = minD2[r];
        v = fminf(v, __shfl_xor(v, 1, 32));
        v = fminf(v, __shfl_xor(v, 2, 32));
        v = fminf(v, __shfl_xor(v, 4, 32));
        v = fminf(v, __shfl_xor(v, 8, 32));
        minD2[r] = v;
    }
    if ((lane & 15) == 0) {                          // lanes 0 and 16
#pragma unroll
        for (int r = 0; r < 8; ++r) {
            const int row = m0 + r + ksel * 8;
            minpart[(size_t)blockIdx.y * BS + row] = minD2[r];
        }
    }
}

// ---------------------------------------------------------------------------
// Kernel 3: combine N-split mins, apply sqrt/margin/relu, mean. One block.
// ---------------------------------------------------------------------------
__global__ __launch_bounds__(256) void finalize_kernel(const float* __restrict__ pd,
                                                       const float* __restrict__ minpart,
                                                       float* __restrict__ out) {
    __shared__ float red[256];
    float s = 0.0f;
    for (int i = threadIdx.x; i < BS; i += 256) {
        float m = minpart[i];
#pragma unroll
        for (int sp = 1; sp < NSPLIT; ++sp)
            m = fminf(m, minpart[(size_t)sp * BS + i]);
        float neg  = sqrtf(fmaxf(m, 0.0f));
        float term = pd[i] - neg + 1.0f;             // MARGIN = 1.0
        s += fmaxf(term, 0.0f);
    }
    red[threadIdx.x] = s;
    __syncthreads();
    for (int step = 128; step > 0; step >>= 1) {
        if (threadIdx.x < step) red[threadIdx.x] += red[threadIdx.x + step];
        __syncthreads();
    }
    if (threadIdx.x == 0) out[0] = red[0] / (float)BS;
}

// ---------------------------------------------------------------------------
extern "C" void kernel_launch(void* const* d_in, const int* in_sizes, int n_in,
                              void* d_out, int out_size, void* d_ws, size_t ws_size,
                              hipStream_t stream) {
    (void)in_sizes; (void)n_in; (void)out_size; (void)ws_size;
    const float* feats = (const float*)d_in[0];
    float* ws   = (float*)d_ws;
    float* sq_a = ws;                           // [BS]
    float* sq_p = ws +  (size_t)BS;             // [BS]
    float* pd   = ws + 2*(size_t)BS;            // [BS]
    float* minp = ws + 3*(size_t)BS;            // [NSPLIT*BS]
    unsigned short* pos_hi = (unsigned short*)(ws + (size_t)(3 + NSPLIT) * BS); // [BS*D]
    unsigned short* pos_lo = pos_hi + (size_t)BS * D;                           // [BS*D]
    float* out  = (float*)d_out;

    prep_kernel<<<BS / 8, 256, 0, stream>>>(feats, sq_a, sq_p, pd, pos_hi, pos_lo);
    tripmin_kernel<<<dim3(BS / 128, NSPLIT), 256, 0, stream>>>(feats, sq_a, sq_p,
                                                               pos_hi, pos_lo, minp);
    finalize_kernel<<<1, 256, 0, stream>>>(pd, minp, out);
}
